// TemplateAwareAttention_68521908240808
// MI455X (gfx1250) — matrix operand order repaired and checked by reading, not compiled, not run on hardware
//
#include <hip/hip_runtime.h>

// MI455X / gfx1250: wave32, WMMA 16x16x32 bf16 (f32 accum).
typedef __attribute__((ext_vector_type(16))) __bf16 v16bf;
typedef __attribute__((ext_vector_type(8)))  float  v8f;
typedef __attribute__((ext_vector_type(4)))  unsigned int u32x4;
typedef unsigned short u16;

#define D_MODEL 768
#define N_HEADS 12
#define HEAD_DIM 64
#define BATCH 4
#define SEQ 2048
#define ROWS (BATCH*SEQ)   // 8192
#define LSG 40             // GEMM LDS row stride (u16): 80B, 16B-aligned, bank-friendly
#define LSV 72             // attention LDS row stride (u16): 144B, 16B-aligned

union Frag {
  v16bf v;
  u32x4 u[2];
  u16   s[16];
};

__device__ __forceinline__ u16 f2bf(float f){
  unsigned u = __float_as_uint(f);
  u += 0x7FFFu + ((u >> 16) & 1u);   // round-to-nearest-even
  return (u16)(u >> 16);
}

__device__ __forceinline__ v8f wmma_bf16(const Frag& a, const Frag& b, v8f c){
  return __builtin_amdgcn_wmma_f32_16x16x32_bf16(false, a.v, false, b.v, (short)0, c, false, false);
}

// ---------------------------------------------------------------------------
// Kernel 0: one-shot fp32 -> bf16 conversion (x and the 4 weight matrices),
// so GEMM inner loops do zero conversion VALU and read half the bytes.
// ---------------------------------------------------------------------------
__global__ __launch_bounds__(256)
void taa_cvt_bf16(const float* __restrict__ src, u16* __restrict__ dst, int n){
  const int i = (blockIdx.x*256 + threadIdx.x)*8;
  if(i >= n) return;
  union{u16 s[8]; u32x4 u;} t;
  #pragma unroll
  for(int j=0;j<8;j++) t.s[j] = f2bf(src[i+j]);
  *(u32x4*)&dst[i] = t.u;
}

// ---------------------------------------------------------------------------
// Kernel 1: fused Q/K/V projection. y = x @ W^T + b.
// Q,K stored bf16 [b,h,s,d] (Q pre-scaled 1/8); V stored bf16 [b,h,d,s]
// (d-major) so attention needs no LDS transpose.
// Block = 128 thr (4 waves); block tile 128x64; wave tile 32x64.
// ---------------------------------------------------------------------------
__global__ __launch_bounds__(128)
void taa_qkv_gemm(const u16* __restrict__ Xb,
                  const u16* __restrict__ Wqb, const float* __restrict__ bq,
                  const u16* __restrict__ Wkb, const float* __restrict__ bk,
                  const u16* __restrict__ Wvb, const float* __restrict__ bv,
                  u16* __restrict__ Qb, u16* __restrict__ Kb, u16* __restrict__ Vb)
{
  __shared__ u16 As[128*LSG];  // 10 KB
  __shared__ u16 Bs[64*LSG];   //  5 KB
  const int tid = threadIdx.x, wave = tid>>5, lane = tid&31;
  const int m0 = blockIdx.x*128, n0 = blockIdx.y*64;
  const int zz = blockIdx.z;
  const u16* W; const float* bias;
  if(zz==0){ W=Wqb; bias=bq; }
  else if(zz==1){ W=Wkb; bias=bk; }
  else { W=Wvb; bias=bv; }

  v8f acc[2][4];
  #pragma unroll
  for(int s_=0;s_<2;s_++) for(int ct=0;ct<4;ct++) for(int i=0;i<8;i++) acc[s_][ct][i]=0.f;

  const int brow = tid>>1, bhf = (tid&1)*16;
  for(int k0=0;k0<D_MODEL;k0+=32){
    __syncthreads();
    { // A fill: one row per thread, 64B copy
      const u16* src = Xb + (size_t)(m0+tid)*D_MODEL + k0;
      #pragma unroll
      for(int c=0;c<4;c++) *(u32x4*)&As[tid*LSG + c*8] = *(const u32x4*)&src[c*8];
    }
    { // B fill: B[n][k] = W[n0+n][k0+k]
      const u16* src = W + (size_t)(n0+brow)*D_MODEL + k0 + bhf;
      *(u32x4*)&Bs[brow*LSG+bhf]   = *(const u32x4*)&src[0];
      *(u32x4*)&Bs[brow*LSG+bhf+8] = *(const u32x4*)&src[8];
    }
    __syncthreads();

    Frag a[2];
    const int kb = (lane>>4)<<3;
    #pragma unroll
    for(int sub=0;sub<2;sub++){
      const int arow = (wave<<5)+(sub<<4)+(lane&15);
      a[sub].u[0] = *(const u32x4*)&As[arow*LSG + kb];
      a[sub].u[1] = *(const u32x4*)&As[arow*LSG + 16 + kb];
    }
    const int kk = (lane>>4)<<3;
    #pragma unroll
    for(int ct=0;ct<4;ct++){
      Frag bf;
      const int col = (ct<<4)+(lane&15);
      bf.u[0] = *(const u32x4*)&Bs[col*LSG + kk];
      bf.u[1] = *(const u32x4*)&Bs[col*LSG + kk + 16];
      #pragma unroll
      for(int sub=0;sub<2;sub++)
        acc[sub][ct] = wmma_bf16(a[sub], bf, acc[sub][ct]);
    }
  }

  // epilogue
  #pragma unroll
  for(int ct=0;ct<4;ct++){
    const int n = n0 + (ct<<4)+(lane&15);
    const int h = n>>6, d = n&63;
    const float bval = bias[n];
    #pragma unroll
    for(int sub=0;sub<2;sub++){
      const int mbase = m0 + (wave<<5)+(sub<<4) + ((lane>>4)<<3);
      const int bb = mbase>>11, sb = mbase&2047;
      if(zz==2){
        // V: [b,h,d,s]; 8 consecutive s per lane -> one b128 store
        union{u16 s[8]; u32x4 u;} t;
        #pragma unroll
        for(int r=0;r<8;r++) t.s[r] = f2bf(acc[sub][ct][r]+bval);
        *(u32x4*)&Vb[(((size_t)bb*N_HEADS+h)*HEAD_DIM + d)*SEQ + sb] = t.u;
      } else {
        u16* dst = (zz==0) ? Qb : Kb;
        const float sc = (zz==0) ? 0.125f : 1.f;
        #pragma unroll
        for(int r=0;r<8;r++)
          dst[(((size_t)bb*N_HEADS+h)*SEQ + (sb+r))*HEAD_DIM + d] = f2bf((acc[sub][ct][r]+bval)*sc);
      }
    }
  }
}

// ---------------------------------------------------------------------------
// Kernel 2: flash attention per (b,h). 4 waves, 64 queries (16/wave).
// Scores computed transposed (keys=M, queries=N): each lane owns one query's
// scores, and Sᵀ's C-layout coincides with the bf16 A-layout needed for P@V.
// ---------------------------------------------------------------------------
__global__ __launch_bounds__(128)
void taa_attention(const u16* __restrict__ Qb, const u16* __restrict__ Kb,
                   const u16* __restrict__ Vb, u16* __restrict__ AO,
                   const int* __restrict__ amask, const int* __restrict__ tids,
                   const float* __restrict__ ttab, const float* __restrict__ alpha)
{
  __shared__ u16 Qs[64*LSV];    // [q][d]
  __shared__ u16 Ks[64*LSV];    // [key][d]
  __shared__ u16 Vt[64*LSV];    // [d][key] (V already d-major in memory)
  __shared__ float maskb[64];

  const int tid=threadIdx.x, wave=tid>>5, lane=tid&31;
  const int bh = blockIdx.y, b = bh/N_HEADS, h = bh - b*N_HEADS;
  const int q0 = blockIdx.x*64;
  const size_t headoff = (size_t)bh*SEQ*HEAD_DIM;
  const float sbias = ttab[tids[b]*N_HEADS + h]*alpha[0];

  { // Q strip -> LDS
    const u16* Qg = Qb + headoff + (size_t)q0*HEAD_DIM;
    for(int i=tid;i<512;i+=128){
      int row=i>>3, c=(i&7)<<3;
      *(u32x4*)&Qs[row*LSV+c] = *(const u32x4*)&Qg[row*64+c];
    }
  }
  __syncthreads();

  // Hoisted Q B-fragments (reused across the whole key loop)
  Frag qfrag[2];
  {
    const int ql = (wave<<4)+(lane&15);
    #pragma unroll
    for(int kd=0;kd<2;kd++){
      const int kk = (kd<<5) + ((lane>>4)<<3);
      qfrag[kd].u[0] = *(const u32x4*)&Qs[ql*LSV + kk];
      qfrag[kd].u[1] = *(const u32x4*)&Qs[ql*LSV + kk + 16];
    }
  }

  v8f o[4];
  #pragma unroll
  for(int ct=0;ct<4;ct++){ for(int i=0;i<8;i++) o[ct][i]=0.f; }
  float m_i = -1e30f, l_i = 0.f;

  for(int kt=0;kt<SEQ;kt+=64){
    __syncthreads();
    {
      const u16* Kg = Kb + headoff + (size_t)kt*HEAD_DIM;
      for(int i=tid;i<512;i+=128){
        int row=i>>3, c=(i&7)<<3;
        *(u32x4*)&Ks[row*LSV+c] = *(const u32x4*)&Kg[row*64+c];
      }
      const u16* Vg = Vb + headoff;   // [d][s]
      for(int i=tid;i<512;i+=128){
        int drow=i>>3, c=(i&7)<<3;
        *(u32x4*)&Vt[drow*LSV+c] = *(const u32x4*)&Vg[(size_t)drow*SEQ + kt + c];
      }
      if(tid<64) maskb[tid] = (amask[b*SEQ+kt+tid]==0) ? -10000.f : 0.f;
    }
    __syncthreads();

    #pragma unroll
    for(int p=0;p<2;p++){              // two 32-key chunks
      v8f sc[2];
      #pragma unroll
      for(int t=0;t<2;t++){            // two 16-key Sᵀ tiles
        #pragma unroll
        for(int i=0;i<8;i++) sc[t][i]=0.f;
        const int krow = (p<<5)+(t<<4)+(lane&15);
        const int kb = (lane>>4)<<3;
        #pragma unroll
        for(int kd=0;kd<2;kd++){
          Frag ka;
          ka.u[0] = *(const u32x4*)&Ks[krow*LSV + (kd<<5) + kb];
          ka.u[1] = *(const u32x4*)&Ks[krow*LSV + (kd<<5) + 16 + kb];
          sc[t] = wmma_bf16(ka, qfrag[kd], sc[t]);
        }
      }
      // template bias + mask, online softmax (lane = one query)
      const int base = (p<<5) + ((lane>>4)<<3);
      float mloc = -1e30f;
      #pragma unroll
      for(int r=0;r<8;r++){
        sc[0][r] += sbias + maskb[base+r];
        sc[1][r] += sbias + maskb[base+16+r];
        mloc = fmaxf(mloc, fmaxf(sc[0][r], sc[1][r]));
      }
      mloc = fmaxf(mloc, __shfl_xor(mloc, 16, 32));
      const float m_new = fmaxf(m_i, mloc);
      const float corr  = __expf(m_i - m_new);
      float rowsum = 0.f;
      Frag pa;                          // P in bf16 A-layout, K=32 keys
      #pragma unroll
      for(int r=0;r<8;r++){
        float p0 = __expf(sc[0][r]-m_new);
        float p1 = __expf(sc[1][r]-m_new);
        rowsum += p0+p1;
        pa.s[r]   = f2bf(p0);
        pa.s[8+r] = f2bf(p1);
      }
      rowsum += __shfl_xor(rowsum, 16, 32);
      l_i = l_i*corr + rowsum;
      m_i = m_new;
      float ce[8];
      #pragma unroll
      for(int r=0;r<8;r++) ce[r] = __shfl(corr, r + ((lane>>4)<<3), 32);
      const int kk = (lane>>4)<<3;
      #pragma unroll
      for(int ct=0;ct<4;ct++){
        #pragma unroll
        for(int r=0;r<8;r++) o[ct][r] *= ce[r];
        Frag vf;
        const int dcol = (ct<<4)+(lane&15);
        vf.u[0] = *(const u32x4*)&Vt[dcol*LSV + (p<<5) + kk];
        vf.u[1] = *(const u32x4*)&Vt[dcol*LSV + (p<<5) + kk + 16];
        o[ct] = wmma_bf16(pa, vf, o[ct]);
      }
    }
  }
  // final 1/l scale + store as [b, s, h*64+d] bf16
  float le[8];
  #pragma unroll
  for(int r=0;r<8;r++){
    float lv = __shfl(l_i, r + ((lane>>4)<<3), 32);
    le[r] = 1.f/fmaxf(lv, 1e-20f);
  }
  #pragma unroll
  for(int ct=0;ct<4;ct++){
    const int dd = h*HEAD_DIM + (ct<<4)+(lane&15);
    #pragma unroll
    for(int r=0;r<8;r++){
      const int srow = q0 + (wave<<4) + r + ((lane>>4)<<3);
      AO[((size_t)b*SEQ + srow)*D_MODEL + dd] = f2bf(o[ct][r]*le[r]);
    }
  }
}

// ---------------------------------------------------------------------------
// Kernel 3: output projection + bias + residual (fp32). 128x64 block tile.
// ---------------------------------------------------------------------------
__global__ __launch_bounds__(128)
void taa_out_gemm(const u16* __restrict__ A, const u16* __restrict__ Wob,
                  const float* __restrict__ bo, const float* __restrict__ X,
                  float* __restrict__ Res)
{
  __shared__ u16 As[128*LSG];
  __shared__ u16 Bs[64*LSG];
  const int tid=threadIdx.x, wave=tid>>5, lane=tid&31;
  const int m0=blockIdx.x*128, n0=blockIdx.y*64;

  v8f acc[2][4];
  #pragma unroll
  for(int s_=0;s_<2;s_++) for(int ct=0;ct<4;ct++) for(int i=0;i<8;i++) acc[s_][ct][i]=0.f;

  const int brow=tid>>1, bhf=(tid&1)*16;
  for(int k0=0;k0<D_MODEL;k0+=32){
    __syncthreads();
    {
      const u16* src = A + (size_t)(m0+tid)*D_MODEL + k0;
      #pragma unroll
      for(int c=0;c<4;c++) *(u32x4*)&As[tid*LSG + c*8] = *(const u32x4*)&src[c*8];
    }
    {
      const u16* src = Wob + (size_t)(n0+brow)*D_MODEL + k0 + bhf;
      *(u32x4*)&Bs[brow*LSG+bhf]   = *(const u32x4*)&src[0];
      *(u32x4*)&Bs[brow*LSG+bhf+8] = *(const u32x4*)&src[8];
    }
    __syncthreads();
    Frag a[2];
    const int kb=(lane>>4)<<3;
    #pragma unroll
    for(int sub=0;sub<2;sub++){
      const int arow=(wave<<5)+(sub<<4)+(lane&15);
      a[sub].u[0] = *(const u32x4*)&As[arow*LSG + kb];
      a[sub].u[1] = *(const u32x4*)&As[arow*LSG + 16 + kb];
    }
    const int kk=(lane>>4)<<3;
    #pragma unroll
    for(int ct=0;ct<4;ct++){
      Frag bf;
      const int col=(ct<<4)+(lane&15);
      bf.u[0] = *(const u32x4*)&Bs[col*LSG + kk];
      bf.u[1] = *(const u32x4*)&Bs[col*LSG + kk + 16];
      #pragma unroll
      for(int sub=0;sub<2;sub++)
        acc[sub][ct] = wmma_bf16(a[sub], bf, acc[sub][ct]);
    }
  }
  #pragma unroll
  for(int ct=0;ct<4;ct++){
    const int n = n0 + (ct<<4)+(lane&15);
    const float bval = bo[n];
    #pragma unroll
    for(int sub=0;sub<2;sub++){
      #pragma unroll
      for(int r=0;r<8;r++){
        const int m = m0 + (wave<<5)+(sub<<4) + r + ((lane>>4)<<3);
        Res[(size_t)m*D_MODEL+n] = acc[sub][ct][r] + bval + X[(size_t)m*D_MODEL+n];
      }
    }
  }
}

// ---------------------------------------------------------------------------
// Kernel 4: LayerNorm, one wave per row.
// ---------------------------------------------------------------------------
__global__ __launch_bounds__(256)
void taa_layernorm(const float* __restrict__ Res, const float* __restrict__ gamma,
                   const float* __restrict__ beta, float* __restrict__ Out)
{
  const int wave = threadIdx.x>>5, lane = threadIdx.x&31;
  const int row = blockIdx.x*8 + wave;
  const float* r = Res + (size_t)row*D_MODEL;
  float v[24], s=0.f, s2=0.f;
  #pragma unroll
  for(int i=0;i<24;i++){ v[i]=r[i*32+lane]; s+=v[i]; s2+=v[i]*v[i]; }
  #pragma unroll
  for(int off=16;off>0;off>>=1){ s+=__shfl_xor(s,off,32); s2+=__shfl_xor(s2,off,32); }
  const float mu = s*(1.f/D_MODEL);
  const float var = s2*(1.f/D_MODEL) - mu*mu;
  const float rstd = rsqrtf(var + 1e-5f);
  float* o = Out + (size_t)row*D_MODEL;
  #pragma unroll
  for(int i=0;i<24;i++){
    const int c=i*32+lane;
    o[c] = (v[i]-mu)*rstd*gamma[c] + beta[c];
  }
}

// ---------------------------------------------------------------------------
extern "C" void kernel_launch(void* const* d_in, const int* in_sizes, int n_in,
                              void* d_out, int out_size, void* d_ws, size_t ws_size,
                              hipStream_t stream) {
  const float* x    = (const float*)d_in[0];
  const int*   tids = (const int*)  d_in[1];
  const int*   am   = (const int*)  d_in[2];
  const float* Wq   = (const float*)d_in[3];  const float* bq = (const float*)d_in[4];
  const float* Wk   = (const float*)d_in[5];  const float* bk = (const float*)d_in[6];
  const float* Wv   = (const float*)d_in[7];  const float* bv = (const float*)d_in[8];
  const float* Wo   = (const float*)d_in[9];  const float* bo = (const float*)d_in[10];
  const float* ttab = (const float*)d_in[11];
  const float* alph = (const float*)d_in[12];
  const float* gam  = (const float*)d_in[13];
  const float* bet  = (const float*)d_in[14];

  // workspace layout (bytes)
  char* ws = (char*)d_ws;
  const size_t XSZ  = (size_t)ROWS*D_MODEL*sizeof(u16);      // 12,582,912
  const size_t WSZ  = (size_t)D_MODEL*D_MODEL*sizeof(u16);   //  1,179,648
  u16*   xb   = (u16*)(ws);
  u16*   Wqb  = (u16*)(ws + XSZ);
  u16*   Wkb  = (u16*)(ws + XSZ +   WSZ);
  u16*   Wvb  = (u16*)(ws + XSZ + 2*WSZ);
  u16*   Wob  = (u16*)(ws + XSZ + 3*WSZ);
  u16*   Qb   = (u16*)(ws + XSZ + 4*WSZ);
  u16*   Kb   = (u16*)(ws + 2*XSZ + 4*WSZ);
  u16*   Vb   = (u16*)(ws + 3*XSZ + 4*WSZ);
  u16*   attn = (u16*)(ws + 4*XSZ + 4*WSZ);
  float* res  = (float*)(ws + 5*XSZ + 4*WSZ);

  const int NX = ROWS*D_MODEL;      // 6,291,456
  const int NW = D_MODEL*D_MODEL;   //   589,824
  taa_cvt_bf16<<<NX/(256*8), 256, 0, stream>>>(x,  xb,  NX);
  taa_cvt_bf16<<<NW/(256*8), 256, 0, stream>>>(Wq, Wqb, NW);
  taa_cvt_bf16<<<NW/(256*8), 256, 0, stream>>>(Wk, Wkb, NW);
  taa_cvt_bf16<<<NW/(256*8), 256, 0, stream>>>(Wv, Wvb, NW);
  taa_cvt_bf16<<<NW/(256*8), 256, 0, stream>>>(Wo, Wob, NW);

  dim3 g1(ROWS/128, D_MODEL/64, 3);
  taa_qkv_gemm<<<g1, 128, 0, stream>>>(xb, Wqb, bq, Wkb, bk, Wvb, bv, Qb, Kb, Vb);

  dim3 g2(SEQ/64, BATCH*N_HEADS);
  taa_attention<<<g2, 128, 0, stream>>>(Qb, Kb, Vb, attn, am, tids, ttab, alph);

  dim3 g3(ROWS/128, D_MODEL/64);
  taa_out_gemm<<<g3, 128, 0, stream>>>(attn, Wob, bo, x, res);

  taa_layernorm<<<ROWS/8, 256, 0, stream>>>(res, gam, bet, (float*)d_out);
}
